// GalerkinSelfAttention_15238543966933
// MI455X (gfx1250) — matrix operand order repaired and checked by reading, not compiled, not run on hardware
//
#include <hip/hip_runtime.h>
#include <hip/hip_bf16.h>

// ---- types ----
typedef __attribute__((ext_vector_type(8)))  __bf16 v8bf;
typedef __attribute__((ext_vector_type(16))) __bf16 v16bf;
typedef __attribute__((ext_vector_type(8)))  float  v8f;
typedef int v4i_gcc __attribute__((vector_size(16)));   // matches builtin param type

#define S_LEN 8192
#define EMB   256
#define HEADS 4
#define HD    64
#define NCH   32            // kv K-chunks per (b,h)
#define CHUNK (S_LEN/NCH)   // 256
#define FSTRIDE 260         // padded fp32 LDS row stride (bank-conflict free)
#define WSTR    264         // padded bf16 LDS row stride (bank-conflict free)

__device__ inline v8f zero8() {
  v8f z;
#pragma unroll
  for (int i = 0; i < 8; i++) z[i] = 0.0f;
  return z;
}

__device__ inline v8bf ld8bf(const __bf16* p) { return *(const v8bf*)p; }

__device__ inline v16bf mk16(v8bf lo, v8bf hi) {
  return __builtin_shufflevector(lo, hi, 0,1,2,3,4,5,6,7,8,9,10,11,12,13,14,15);
}

__device__ inline v8bf cvt8(const float* p) {
  v8bf r;
#pragma unroll
  for (int i = 0; i < 8; i++) r[i] = (__bf16)p[i];
  return r;
}

__device__ inline v8f wmma_bf16(v16bf a, v16bf b, v8f c) {
  // D = A(16x32 bf16) x B(32x16 bf16) + C(16x16 f32)
  return __builtin_amdgcn_wmma_f32_16x16x32_bf16(false, a, false, b, (short)0, c, false, false);
}

// ---- CDNA5 async global->LDS copy (probe: builtin exists; param0 = int4 AS1*) ----
__device__ inline void async_cp16(const __bf16* g, __bf16* l) {
#if __has_builtin(__builtin_amdgcn_global_load_async_to_lds_b128)
  __builtin_amdgcn_global_load_async_to_lds_b128(
      (__attribute__((address_space(1))) v4i_gcc*)(uintptr_t)g,
      (__attribute__((address_space(3))) v4i_gcc*)(uintptr_t)l, 0, 0);
#else
  *(v8bf*)l = *(const v8bf*)g;
#endif
}

__device__ inline void async_wait() {
#if __has_builtin(__builtin_amdgcn_global_load_async_to_lds_b128)
# if __has_builtin(__builtin_amdgcn_s_wait_asynccnt)
  __builtin_amdgcn_s_wait_asynccnt(0);
# else
  asm volatile("s_wait_asynccnt 0" ::: "memory");
# endif
#endif
}

// ---------------------------------------------------------------------------
// Kernel 0: convert the four 256x256 fp32 weight matrices to bf16 scratch.
// Layout: [wq | wk | wv | wo], each 65536 elements, row-major (= B^T operand).
// ---------------------------------------------------------------------------
__global__ void k_cvt_w(const float* __restrict__ wq, const float* __restrict__ wk,
                        const float* __restrict__ wv, const float* __restrict__ wo,
                        __bf16* __restrict__ dst) {
  int i = blockIdx.x * blockDim.x + threadIdx.x;   // 0..65535
  dst[i]           = (__bf16)wq[i];
  dst[i +  65536]  = (__bf16)wk[i];
  dst[i + 131072]  = (__bf16)wv[i];
  dst[i + 196608]  = (__bf16)wo[i];
}

// ---------------------------------------------------------------------------
// Kernel 1: fused QKV projection + LayerNorm(k,v) + layout transform.
// Block = 128 threads (4 waves); each wave owns TWO 16-row strips (32 rows),
// so every B-operand tile load feeds two WMMAs (halves B traffic).
//   q -> qbuf [B,H,S,64]  row-major (A-operand friendly for kernel 3)
//   k,v -> kT,vT [B,H,64,S]  transposed (A/B-operand friendly for kernel 2)
// ---------------------------------------------------------------------------
__global__ __launch_bounds__(128) void k_qkv(
    const float* __restrict__ seq, const __bf16* __restrict__ wB,
    const float* __restrict__ g_k, const float* __restrict__ b_k,
    const float* __restrict__ g_v, const float* __restrict__ b_v,
    __bf16* __restrict__ qbuf, __bf16* __restrict__ kT, __bf16* __restrict__ vT) {

  __shared__ __align__(16) float strip[4][2][16 * FSTRIDE];  // 133 KB
  __shared__ float s_mu[4][2][16][4];
  __shared__ float s_rs[4][2][16][4];

  const int w    = threadIdx.x >> 5;
  const int lane = threadIdx.x & 31;
  const int row  = lane & 15;
  const int hi   = lane >> 4;

  const int blk = blockIdx.x;              // 256 blocks: b*64 + blk128
  const int b   = blk >> 6;
  const int s0  = (blk & 63) * 128 + w * 32;   // this wave's first seq row

  // Load two A-strips: seq rows [s0..s0+31] x 256, fp32 -> bf16, WMMA A layout.
  v16bf a[2][8];
#pragma unroll
  for (int st = 0; st < 2; st++) {
    const float* srow = seq + ((size_t)(b * S_LEN + s0 + st * 16 + row)) * EMB;
#pragma unroll
    for (int c = 0; c < 8; c++) {
      v8bf lo = cvt8(srow + c * 32 + hi * 8);
      v8bf h8 = cvt8(srow + c * 32 + hi * 8 + 16);
      a[st][c] = mk16(lo, h8);
    }
  }

  for (int m = 0; m < 3; m++) {                    // 0:q 1:k 2:v
    const __bf16* W = wB + m * 65536;              // row-major = B^T operand
    for (int nt = 0; nt < 16; nt++) {              // 16 output col tiles
      v8f acc0 = zero8(), acc1 = zero8();
#pragma unroll
      for (int c = 0; c < 8; c++) {                // K = 256 in steps of 32
        const __bf16* bp = W + (size_t)(nt * 16 + row) * 256 + c * 32 + hi * 8;
        v16bf Bv = mk16(ld8bf(bp), ld8bf(bp + 16));
        acc0 = wmma_bf16(a[0][c], Bv, acc0);
        acc1 = wmma_bf16(a[1][c], Bv, acc1);
      }
#pragma unroll
      for (int r = 0; r < 8; r++) {
        strip[w][0][(hi * 8 + r) * FSTRIDE + nt * 16 + row] = acc0[r];
        strip[w][1][(hi * 8 + r) * FSTRIDE + nt * 16 + row] = acc1[r];
      }
    }

    for (int st = 0; st < 2; st++) {
      const int sb = s0 + st * 16;
      if (m == 0) {
        // q: straight row-major bf16 store, per (row,head) pair.
#pragma unroll
        for (int pi = 0; pi < 2; pi++) {
          int p = lane + pi * 32;                  // 0..63
          int r = p & 15, h = p >> 4;
          const float* src = &strip[w][st][r * FSTRIDE + h * 64];
          __bf16* dst = qbuf + (((size_t)(b * HEADS + h)) * S_LEN + (sb + r)) * HD;
#pragma unroll
          for (int j = 0; j < 8; j++) *(v8bf*)(dst + j * 8) = cvt8(src + j * 8);
        }
      } else {
        // LayerNorm stats per (row, head)
#pragma unroll
        for (int pi = 0; pi < 2; pi++) {
          int p = lane + pi * 32;
          int r = p & 15, h = p >> 4;
          const float* src = &strip[w][st][r * FSTRIDE + h * 64];
          float s = 0.f;
          for (int d = 0; d < HD; d++) s += src[d];
          float mu = s * (1.0f / HD);
          float vr = 0.f;
          for (int d = 0; d < HD; d++) { float t = src[d] - mu; vr += t * t; }
          s_mu[w][st][r][h] = mu;
          s_rs[w][st][r][h] = rsqrtf(vr * (1.0f / HD) + 1e-5f);
        }
        const float* gg  = (m == 1) ? g_k : g_v;
        const float* bbv = (m == 1) ? b_k : b_v;
        __bf16* dstT = (m == 1) ? kT : vT;
        // transposed store: each lane handles 8 columns, 16 contiguous s each
        for (int i = 0; i < 8; i++) {
          int col = lane + i * 32;                 // 0..255
          int h = col >> 6, d = col & 63;
          float gd = gg[d], bd = bbv[d];
          v8bf t0, t1;
#pragma unroll
          for (int r = 0; r < 8; r++) {
            t0[r] = (__bf16)((strip[w][st][r * FSTRIDE + col] - s_mu[w][st][r][h]) *
                             s_rs[w][st][r][h] * gd + bd);
            t1[r] = (__bf16)((strip[w][st][(r + 8) * FSTRIDE + col] - s_mu[w][st][r + 8][h]) *
                             s_rs[w][st][r + 8][h] * gd + bd);
          }
          __bf16* dp = dstT + (((size_t)(b * HEADS + h)) * HD + d) * S_LEN + sb;
          *(v8bf*)dp = t0;
          *(v8bf*)(dp + 8) = t1;
        }
      }
    }
  }
}

// ---------------------------------------------------------------------------
// Kernel 2: kv partials. One block per (b,h,chunk); wave w owns d-tile w.
// kv[d,e] = sum_s kT[d,s] * vT[e,s]^T ; A = kT rows, B^T = vT rows (both
// contiguous). Partials stored transposed (lane-contiguous in d).
// ---------------------------------------------------------------------------
__global__ __launch_bounds__(128) void k_kv(const __bf16* __restrict__ kT,
                                            const __bf16* __restrict__ vT,
                                            float* __restrict__ partT) {
  const int w    = threadIdx.x >> 5;
  const int lane = threadIdx.x & 31;
  const int row  = lane & 15;
  const int hi   = lane >> 4;

  const int blk = blockIdx.x;          // bh*NCH + ch
  const int bh  = blk / NCH;
  const int ch  = blk % NCH;

  const __bf16* kbase = kT + (size_t)bh * HD * S_LEN;
  const __bf16* vbase = vT + (size_t)bh * HD * S_LEN;

  v8f acc[4] = {zero8(), zero8(), zero8(), zero8()};
  const int sb0 = ch * CHUNK;
  for (int sb = sb0; sb < sb0 + CHUNK; sb += 32) {
    const __bf16* ap = kbase + (size_t)(w * 16 + row) * S_LEN + sb + hi * 8;
    v16bf A = mk16(ld8bf(ap), ld8bf(ap + 16));
#pragma unroll
    for (int et = 0; et < 4; et++) {
      const __bf16* bp = vbase + (size_t)(et * 16 + row) * S_LEN + sb + hi * 8;
      acc[et] = wmma_bf16(A, mk16(ld8bf(bp), ld8bf(bp + 16)), acc[et]);
    }
  }
  float* part = partT + (size_t)blk * HD * HD;   // [e][d] transposed
#pragma unroll
  for (int et = 0; et < 4; et++) {
    int e = et * 16 + row;
#pragma unroll
    for (int r = 0; r < 8; r++)
      part[e * HD + w * 16 + hi * 8 + r] = acc[et][r];
  }
}

// ---------------------------------------------------------------------------
// Kernel 2b: reduce partials over chunks, scale by 1/S, emit kvT bf16
// [B,H,64(e),64(d)] — exactly the B^T operand layout for kernel 3.
// ---------------------------------------------------------------------------
__global__ void k_kvred(const float* __restrict__ partT, __bf16* __restrict__ kvT) {
  int i = blockIdx.x * blockDim.x + threadIdx.x;  // 0..65535
  int bh = i >> 12, ed = i & 4095;
  float s = 0.f;
#pragma unroll 4
  for (int ch = 0; ch < NCH; ch++)
    s += partT[((size_t)(bh * NCH + ch)) * 4096 + ed];
  kvT[i] = (__bf16)(s * (1.0f / S_LEN));
}

// ---------------------------------------------------------------------------
// Kernel 3: ga = q @ kv (per head) -> LDS bf16 strip -> out = ga @ wo^T + bo.
// Block = 128 threads (4 waves); wave = 16 rows. wo is staged block-wide in
// LDS (two 128-row halves) via the CDNA5 async global->LDS path.
// ---------------------------------------------------------------------------
__global__ __launch_bounds__(128) void k_out(const __bf16* __restrict__ qbuf,
                                             const __bf16* __restrict__ kvT,
                                             const __bf16* __restrict__ woB,
                                             const float* __restrict__ bo,
                                             float* __restrict__ out) {
  __shared__ __align__(16) __bf16 wolds[128 * WSTR];     // 66 KB (half of wo)
  __shared__ __align__(16) __bf16 gasl[4][16 * WSTR];    // 33 KB ga strips

  const int t    = threadIdx.x;
  const int w    = t >> 5;
  const int lane = t & 31;
  const int row  = lane & 15;
  const int hi   = lane >> 4;

  const int blk = blockIdx.x;
  const int b   = blk >> 7;
  const int s0  = (blk & 127) * 64 + w * 16;

  // Hint: pull second half of wo toward L2 while we work on the first.
  __builtin_prefetch(woB + 128 * 256, 0, 1);

  // Stage 1: ga strips per head
  for (int h = 0; h < HEADS; h++) {
    const __bf16* qrow = qbuf + (((size_t)(b * HEADS + h)) * S_LEN + s0 + row) * HD;
    v16bf aq[2];
#pragma unroll
    for (int c = 0; c < 2; c++) {
      const __bf16* p = qrow + c * 32 + hi * 8;
      aq[c] = mk16(ld8bf(p), ld8bf(p + 16));
    }
    const __bf16* kvb = kvT + (size_t)(b * HEADS + h) * 4096;
#pragma unroll
    for (int et = 0; et < 4; et++) {
      v8f acc = zero8();
#pragma unroll
      for (int c = 0; c < 2; c++) {
        const __bf16* bp = kvb + (et * 16 + row) * HD + c * 32 + hi * 8;
        acc = wmma_bf16(aq[c], mk16(ld8bf(bp), ld8bf(bp + 16)), acc);
      }
#pragma unroll
      for (int r = 0; r < 8; r++)
        gasl[w][(hi * 8 + r) * WSTR + h * 64 + et * 16 + row] = (__bf16)acc[r];
    }
  }

  // A-operand strips from LDS (per-wave data; no cross-wave hazard yet)
  v16bf ag[8];
#pragma unroll
  for (int c = 0; c < 8; c++) {
    const __bf16* p = &gasl[w][row * WSTR + c * 32 + hi * 8];
    ag[c] = mk16(*(const v8bf*)p, *(const v8bf*)(p + 16));
  }

  // Stage 2: out projection over K=256, wo staged in LDS half-matrix at a time
  for (int half = 0; half < 2; half++) {
    __syncthreads();                     // all waves done with previous wolds
#pragma unroll 4
    for (int p = 0; p < 32; p++) {       // 4096 16B chunks / 128 threads
      int k  = p * 128 + t;
      int r  = k >> 5;                   // 0..127 local row
      int co = (k & 31) * 8;             // element offset in row
      async_cp16(woB + (size_t)(half * 128 + r) * 256 + co, &wolds[r * WSTR + co]);
    }
    async_wait();
    __syncthreads();                     // wolds visible to all waves

    for (int nt = 0; nt < 8; nt++) {
      v8f acc = zero8();
#pragma unroll
      for (int c = 0; c < 8; c++) {
        const __bf16* bp = &wolds[(nt * 16 + row) * WSTR + c * 32 + hi * 8];
        acc = wmma_bf16(ag[c], mk16(ld8bf(bp), ld8bf(bp + 16)), acc);
      }
      int n = (half * 8 + nt) * 16 + row;
      float bias = bo[n];
#pragma unroll
      for (int r = 0; r < 8; r++)
        out[((size_t)(b * S_LEN) + s0 + hi * 8 + r) * EMB + n] = acc[r] + bias;
    }
  }
}

// ---------------------------------------------------------------------------
extern "C" void kernel_launch(void* const* d_in, const int* in_sizes, int n_in,
                              void* d_out, int out_size, void* d_ws, size_t ws_size,
                              hipStream_t stream) {
  const float* seq    = (const float*)d_in[0];
  const float* wq     = (const float*)d_in[1];
  const float* wk     = (const float*)d_in[2];
  const float* wv     = (const float*)d_in[3];
  const float* wo     = (const float*)d_in[4];
  const float* bo     = (const float*)d_in[5];
  const float* ln_k_g = (const float*)d_in[6];
  const float* ln_k_b = (const float*)d_in[7];
  const float* ln_v_g = (const float*)d_in[8];
  const float* ln_v_b = (const float*)d_in[9];

  char* ws = (char*)d_ws;
  __bf16* wB    = (__bf16*)ws;                               // 4*65536 bf16 = 512 KB
  __bf16* qbuf  = (__bf16*)(ws + 524288);                    // 16 MB
  __bf16* kT    = (__bf16*)(ws + 524288 + 16777216);         // 16 MB
  __bf16* vT    = (__bf16*)(ws + 524288 + 2 * 16777216);     // 16 MB
  float*  partT = (float*)(ws + 524288 + 3 * 16777216);      // 16*32*4096*4 = 8 MB
  __bf16* kvT   = (__bf16*)(ws + 524288 + 3 * 16777216 + 8388608);  // 128 KB

  k_cvt_w<<<256, 256, 0, stream>>>(wq, wk, wv, wo, wB);
  k_qkv<<<256, 128, 0, stream>>>(seq, wB, ln_k_g, ln_k_b, ln_v_g, ln_v_b, qbuf, kT, vT);
  k_kv<<<HEADS * 4 * NCH, 128, 0, stream>>>(kT, vT, partT);
  k_kvred<<<256, 256, 0, stream>>>(partT, kvT);
  k_out<<<512, 128, 0, stream>>>(qbuf, kvT, wB + 3 * 65536, bo, (float*)d_out);
}
